// RelativeSelfAttention_46110768890096
// MI455X (gfx1250) — matrix operand order repaired and checked by reading, not compiled, not run on hardware
//
#include <hip/hip_runtime.h>
#include <hip/hip_bf16.h>
#include <stdint.h>

typedef __attribute__((ext_vector_type(16))) _Float16 v16h;
typedef __attribute__((ext_vector_type(8)))  float    v8f;

#define HD 12
#define DH 64
#define NSEQ 1024
#define CC 768

union Frag { v16h h; uint32_t u[8]; };

// ---------------------------------------------------------------- convert
__global__ void cvt_f32_f16(const float* __restrict__ s, _Float16* __restrict__ d, int n) {
  int i = blockIdx.x * 256 + threadIdx.x;
  if (i < n) d[i] = (_Float16)s[i];
}

// ---------------------------------------------------------------- GEMM
// C[m,n] = sum_k A[m,k] * Bw[n,k]   (A: Mx768 f16 row-major, Bw: Nx768 f16 row-major)
// MODE 0: QKV epilogue -> scatter into Qh (scaled), Kh, Vt(transposed)
// MODE 1: proj epilogue -> f32 out + bias
template<int MODE>
__global__ __launch_bounds__(256) void gemm16(
    const _Float16* __restrict__ A, const _Float16* __restrict__ Bw,
    _Float16* __restrict__ Qh, _Float16* __restrict__ Kh, _Float16* __restrict__ Vt,
    const float* __restrict__ bias, float* __restrict__ Out)
{
  __shared__ uint32_t lds_a[64 * 17];
  __shared__ uint32_t lds_b[64 * 17];
  const int lane = threadIdx.x & 31;
  const int wid  = threadIdx.x >> 5;
  const int wm = wid >> 1, wn = wid & 1;           // 4x2 wave grid -> 64x64 tile
  const int m0 = blockIdx.y * 64;
  const int n0 = blockIdx.x * 64;

  v8f acc[2] = {};
  for (int k0 = 0; k0 < CC; k0 += 32) {
#pragma unroll
    for (int i = 0; i < 4; ++i) {                  // stage 64x32 f16 A and B tiles
      int idx = threadIdx.x + 256 * i;
      int r = idx >> 4, cp = idx & 15;
      lds_a[r * 17 + cp] = *(const uint32_t*)(A  + (size_t)(m0 + r) * CC + k0 + 2 * cp);
      lds_b[r * 17 + cp] = *(const uint32_t*)(Bw + (size_t)(n0 + r) * CC + k0 + 2 * cp);
      if (k0 + 32 < CC) {                          // prefetch next K tile
        __builtin_prefetch(A  + (size_t)(m0 + r) * CC + k0 + 32 + 2 * cp, 0, 0);
        __builtin_prefetch(Bw + (size_t)(n0 + r) * CC + k0 + 32 + 2 * cp, 0, 0);
      }
    }
    __syncthreads();
    Frag a, b0, b1;
#pragma unroll
    for (int v = 0; v < 8; ++v)                    // A fragment (16x32), ISA striping
      a.u[v] = lds_a[(wm * 16 + (lane & 15)) * 17 + 8 * (v >> 2) + 4 * (lane >> 4) + (v & 3)];
#pragma unroll
    for (int v = 0; v < 8; ++v) {                  // B fragments (32x16)
      b0.u[v] = lds_b[(wn * 32 +      (lane & 15)) * 17 + 8 * (v >> 2) + 4 * (lane >> 4) + (v & 3)];
      b1.u[v] = lds_b[(wn * 32 + 16 + (lane & 15)) * 17 + 8 * (v >> 2) + 4 * (lane >> 4) + (v & 3)];
    }
    acc[0] = __builtin_amdgcn_wmma_f32_16x16x32_f16(false, a.h, false, b0.h, (short)0, acc[0], false, false);
    acc[1] = __builtin_amdgcn_wmma_f32_16x16x32_f16(false, a.h, false, b1.h, (short)0, acc[1], false, false);
    __syncthreads();
  }
#pragma unroll
  for (int t = 0; t < 2; ++t) {
#pragma unroll
    for (int r = 0; r < 8; ++r) {
      int m = m0 + wm * 16 + r + 8 * (lane >> 4);  // C/D layout: vgpr r, lane half
      int n = n0 + wn * 32 + t * 16 + (lane & 15);
      float val = acc[t][r];
      if (MODE == 0) {
        int which = n / CC; int c = n - which * CC;
        int hh = c >> 6, dd = c & 63;
        int bb = m >> 10, nn = m & (NSEQ - 1);
        if (which == 0)
          Qh[(((size_t)bb * HD + hh) * NSEQ + nn) * DH + dd] = (_Float16)(val * 0.125f);
        else if (which == 1)
          Kh[(((size_t)bb * HD + hh) * NSEQ + nn) * DH + dd] = (_Float16)val;
        else
          Vt[(((size_t)bb * HD + hh) * DH + dd) * NSEQ + nn] = (_Float16)val;
      } else {
        Out[(size_t)m * CC + n] = val + bias[n];
      }
    }
  }
}

// ---------------------------------------------------------------- attention
// One wave handles one (b,h) 16-row query tile; flash-style over 64-key chunks.
__global__ __launch_bounds__(128) void attn16(
    const _Float16* __restrict__ Qh, const _Float16* __restrict__ Kh,
    const _Float16* __restrict__ Vt, const float* __restrict__ rel_table,
    _Float16* __restrict__ Ah)
{
  __shared__ float relc[132];
  __shared__ __align__(16) _Float16 ldsp[4][16][66];   // per-wave P restripe buffer
  const int lane = threadIdx.x & 31;
  const int wid  = threadIdx.x >> 5;
  const int wg = blockIdx.x * 4 + wid;
  const int qt = wg & 63;
  const int h  = (wg >> 6) % HD;                   // uniform across the block
  const int b  = wg / (64 * HD);
  const int q0 = qt * 16;
  const int bh = b * HD + h;

  for (int t = threadIdx.x; t < 129; t += 128)
    relc[t] = rel_table[t * HD + h];
  __syncthreads();

  Frag aq0, aq1;                                   // Q tile 16x64 as two A fragments
  {
    const uint32_t* qrow = (const uint32_t*)(Qh + ((size_t)bh * NSEQ + q0 + (lane & 15)) * DH);
#pragma unroll
    for (int v = 0; v < 8; ++v) {
      int bi = 8 * (v >> 2) + 4 * (lane >> 4) + (v & 3);
      aq0.u[v] = qrow[bi];
      aq1.u[v] = qrow[16 + bi];
    }
  }

  v8f acc_o[4] = {};
  float rmax[8], rsum[8];
#pragma unroll
  for (int r = 0; r < 8; ++r) { rmax[r] = -1e30f; rsum[r] = 0.f; }

  for (int kc = 0; kc < NSEQ; kc += 64) {
    v8f s[4] = {};
#pragma unroll
    for (int t = 0; t < 4; ++t) {                  // S = Q K^T (16x64 this chunk)
      const uint32_t* krow = (const uint32_t*)(Kh + ((size_t)bh * NSEQ + kc + 16 * t + (lane & 15)) * DH);
      Frag bk0, bk1;
#pragma unroll
      for (int v = 0; v < 8; ++v) {
        bk0.u[v] = krow[8 * (v >> 2) + 4 * (lane >> 4) + (v & 3)];
        bk1.u[v] = krow[16 + 8 * (v >> 2) + 4 * (lane >> 4) + (v & 3)];
      }
      s[t] = __builtin_amdgcn_wmma_f32_16x16x32_f16(false, aq0.h, false, bk0.h, (short)0, s[t], false, false);
      s[t] = __builtin_amdgcn_wmma_f32_16x16x32_f16(false, aq1.h, false, bk1.h, (short)0, s[t], false, false);
    }
    const int iRow = q0 + 8 * (lane >> 4);
#pragma unroll
    for (int t = 0; t < 4; ++t) {                  // + relative position bias
      int j = kc + 16 * t + (lane & 15);
#pragma unroll
      for (int r = 0; r < 8; ++r) {
        int idx = j - (iRow + r) + 64;
        idx = idx < 0 ? 0 : (idx > 128 ? 128 : idx);
        s[t][r] += relc[idx];
      }
    }
    float scl[8];
#pragma unroll
    for (int r = 0; r < 8; ++r) {                  // online-softmax row max (16-lane groups)
      float mx = fmaxf(fmaxf(s[0][r], s[1][r]), fmaxf(s[2][r], s[3][r]));
      mx = fmaxf(mx, __shfl_xor(mx, 1));
      mx = fmaxf(mx, __shfl_xor(mx, 2));
      mx = fmaxf(mx, __shfl_xor(mx, 4));
      mx = fmaxf(mx, __shfl_xor(mx, 8));
      float nm = fmaxf(rmax[r], mx);
      scl[r] = __expf(rmax[r] - nm);
      rmax[r] = nm;
    }
    float lsum[8];
#pragma unroll
    for (int r = 0; r < 8; ++r) lsum[r] = 0.f;
#pragma unroll
    for (int t = 0; t < 4; ++t) {                  // P = exp(S-max): LDS restripe C->A layout
#pragma unroll
      for (int r = 0; r < 8; ++r) {
        float p = __expf(s[t][r] - rmax[r]);
        lsum[r] += p;
        ldsp[wid][r + 8 * (lane >> 4)][16 * t + (lane & 15)] = (_Float16)p;
      }
    }
#pragma unroll
    for (int r = 0; r < 8; ++r) {
      float sm = lsum[r];
      sm += __shfl_xor(sm, 1);
      sm += __shfl_xor(sm, 2);
      sm += __shfl_xor(sm, 4);
      sm += __shfl_xor(sm, 8);
      rsum[r] = rsum[r] * scl[r] + sm;
#pragma unroll
      for (int tO = 0; tO < 4; ++tO)
        acc_o[tO][r] *= scl[r];
    }
    asm volatile("s_wait_dscnt 0" ::: "memory");   // P stores visible to own wave's loads
    Frag ap0, ap1;
    {
      const uint32_t* prow = (const uint32_t*)&ldsp[wid][lane & 15][0];
#pragma unroll
      for (int v = 0; v < 8; ++v) {
        int bi = 8 * (v >> 2) + 4 * (lane >> 4) + (v & 3);
        ap0.u[v] = prow[bi];
        ap1.u[v] = prow[16 + bi];
      }
    }
#pragma unroll
    for (int tO = 0; tO < 4; ++tO) {               // O += P V  (V pre-transposed: rows = d)
      const uint32_t* vrow = (const uint32_t*)(Vt + ((size_t)bh * DH + 16 * tO + (lane & 15)) * NSEQ + kc);
      Frag bv0, bv1;
#pragma unroll
      for (int v = 0; v < 8; ++v) {
        bv0.u[v] = vrow[8 * (v >> 2) + 4 * (lane >> 4) + (v & 3)];
        bv1.u[v] = vrow[16 + 8 * (v >> 2) + 4 * (lane >> 4) + (v & 3)];
      }
      acc_o[tO] = __builtin_amdgcn_wmma_f32_16x16x32_f16(false, ap0.h, false, bv0.h, (short)0, acc_o[tO], false, false);
      acc_o[tO] = __builtin_amdgcn_wmma_f32_16x16x32_f16(false, ap1.h, false, bv1.h, (short)0, acc_o[tO], false, false);
    }
    asm volatile("s_wait_dscnt 0" ::: "memory");   // drain P reads before next overwrite
  }
#pragma unroll
  for (int tO = 0; tO < 4; ++tO) {                 // write attn out (B,N,C) f16
#pragma unroll
    for (int r = 0; r < 8; ++r) {
      int m = q0 + r + 8 * (lane >> 4);
      int d = 16 * tO + (lane & 15);
      float val = acc_o[tO][r] / rsum[r];
      Ah[((size_t)b * NSEQ + m) * CC + h * DH + d] = (_Float16)val;
    }
  }
}

// ---------------------------------------------------------------- launch
extern "C" void kernel_launch(void* const* d_in, const int* in_sizes, int n_in,
                              void* d_out, int out_size, void* d_ws, size_t ws_size,
                              hipStream_t stream) {
  const float* x      = (const float*)d_in[0];
  const float* qkv_w  = (const float*)d_in[1];
  const float* proj_w = (const float*)d_in[2];
  const float* proj_b = (const float*)d_in[3];
  const float* rel    = (const float*)d_in[4];

  const size_t M = 4096;                   // B*N
  _Float16* xh  = (_Float16*)d_ws;
  _Float16* wqk = xh  + M * CC;            // 3,145,728
  _Float16* wpr = wqk + (size_t)2304 * CC; // 1,769,472
  _Float16* Qh  = wpr + (size_t)CC * CC;   //   589,824
  _Float16* Kh  = Qh  + (size_t)48 * NSEQ * DH;
  _Float16* Vt  = Kh  + (size_t)48 * NSEQ * DH;
  _Float16* Ah  = Vt  + (size_t)48 * NSEQ * DH;
  float* Out = (float*)d_out;

  cvt_f32_f16<<<(int)((M * CC + 255) / 256), 256, 0, stream>>>(x, xh, (int)(M * CC));
  cvt_f32_f16<<<(2304 * CC + 255) / 256, 256, 0, stream>>>(qkv_w, wqk, 2304 * CC);
  cvt_f32_f16<<<(CC * CC + 255) / 256, 256, 0, stream>>>(proj_w, wpr, CC * CC);

  gemm16<0><<<dim3(2304 / 64, 4096 / 64), 256, 0, stream>>>(xh, wqk, Qh, Kh, Vt, nullptr, nullptr);
  attn16<<<dim3(4 * HD * 64 / 4), 128, 0, stream>>>(Qh, Kh, Vt, rel, Ah);
  gemm16<1><<<dim3(CC / 64, 4096 / 64), 256, 0, stream>>>(Ah, wpr, nullptr, nullptr, nullptr, proj_b, Out);
}
